// MPTAttention_54589034332512
// MI455X (gfx1250) — compile-verified, matrix operand order checked by reading, not run on hardware
//
#include <hip/hip_runtime.h>

typedef __attribute__((ext_vector_type(16))) _Float16 v16h;
typedef __attribute__((ext_vector_type(8)))  float    v8f;

#define S_LEN   2048
#define DMODEL  2048
#define NHEADS  16
#define DHEAD   128
#define LD_QKV  (3 * DMODEL)   // 6144

#define LDS_LDA 40             // 32 f16 + 8 f16 pad = 80B rows -> conflict-free b128 reads

union FragU { uint4 u[2]; v16h h; };
union AccU  { v8f v; float f[8]; };

__device__ __forceinline__ v8f wmma16(v16h a, v16h b, v8f c) {
    // v_wmma_f32_16x16x32_f16  (args: neg_a, A, neg_b, B, c_mod, C, reuse_a, reuse_b)
    return __builtin_amdgcn_wmma_f32_16x16x32_f16(false, a, false, b, (short)0, c, false, false);
}

// Load a 16x32 f16 fragment (A layout; the B-fragment holds one column per lane
// with the same K packing, so the same loader serves both).
// ISA 7.12.2: lanes 0-15 rows 0-15; VGPR0-3 hold K = half*8 + 0..7,
// VGPR4-7 hold K = 16 + half*8 + 0..7 (half = lane>>4).
__device__ __forceinline__ v16h load_frag_rm(const _Float16* base, int ld, int k0) {
    const int lane = threadIdx.x & 31;
    const int row  = lane & 15;
    const int lh   = lane >> 4;
    const _Float16* p = base + (size_t)row * ld + (k0 + lh * 8);
    FragU f;
    f.u[0] = *(const uint4*)(p);        // K = k0 + lh*8 .. +7
    f.u[1] = *(const uint4*)(p + 16);   // K = k0 + 16 + lh*8 .. +7
    return f.h;
}

// ---- CDNA5 async copy: global memory -> LDS, tracked by ASYNCcnt ------------
__device__ __forceinline__ void async_b128_to_lds(const _Float16* g, _Float16* lds_p) {
    unsigned lds_addr = (unsigned)(size_t)lds_p;           // low 32 bits = LDS offset
    unsigned long long ga = (unsigned long long)(size_t)g;
    asm volatile("global_load_async_to_lds_b128 %0, %1, off"
                 :: "v"(lds_addr), "v"(ga) : "memory");
}
__device__ __forceinline__ void wait_async0() {
    asm volatile("s_wait_asynccnt 0" ::: "memory");
}

// ---------------------------------------------------------------- conversion
__global__ void cvt_f32_to_f16(const float* __restrict__ in, _Float16* __restrict__ out, int n) {
    int i = blockIdx.x * blockDim.x + threadIdx.x;
    int stride = gridDim.x * blockDim.x;
    for (; i < n; i += stride) out[i] = (_Float16)in[i];
}

// ----------------------------------------------------- V transpose to [H*d, S]
__global__ void transpose_v(const _Float16* __restrict__ qkv, _Float16* __restrict__ vt) {
    int idx = blockIdx.x * blockDim.x + threadIdx.x;      // over NHEADS*DHEAD*S_LEN
    int s   = idx & (S_LEN - 1);
    int hd  = idx >> 11;
    if (hd < NHEADS * DHEAD)
        vt[(size_t)hd * S_LEN + s] = qkv[(size_t)s * LD_QKV + 2 * DMODEL + hd];
}

// --------------------------------------------------------------- GEMM A * W^T
// A [M,K] f16 row-major, W [N,K] f16 row-major. Workgroup macro-tile 128x128,
// 8 waves each computing 32x64. Tiles stream into LDS via async copies with
// double buffering; fragments are fed from LDS (conflict-free ds_load_b128).
template <int OUTF16>
__global__ __launch_bounds__(256) void gemm_lds(const _Float16* __restrict__ A,
                                                const _Float16* __restrict__ W,
                                                void* __restrict__ Cout,
                                                int M, int N, int K) {
    __shared__ __align__(16) _Float16 Asm[2][128][LDS_LDA];
    __shared__ __align__(16) _Float16 Bsm[2][128][LDS_LDA];

    const int wid  = threadIdx.x >> 5;
    const int wm   = wid & 3;                 // 4 m-slots of 32 rows
    const int wn   = wid >> 2;                // 2 n-slots of 64 cols
    const int mblk = blockIdx.y;              // 128-row macro tile
    const int nblk = blockIdx.x;              // 128-col macro tile

    const _Float16* Ag = A + (size_t)(mblk * 128) * K;
    const _Float16* Wg = W + (size_t)(nblk * 128) * K;

    // Each thread carries 2 b128 chunks of A and 2 of B per stage.
    const int t   = threadIdx.x;
    const int r0  = t >> 2;                   // rows 0..63
    const int cq0 = (t & 3) * 8;              // 16B chunk within a 32-f16 row

    auto issue_stage = [&](int k0, int buf) {
#pragma unroll
        for (int i = 0; i < 2; ++i) {
            int row = r0 + i * 64;
            async_b128_to_lds(Ag + (size_t)row * K + k0 + cq0, &Asm[buf][row][cq0]);
            async_b128_to_lds(Wg + (size_t)row * K + k0 + cq0, &Bsm[buf][row][cq0]);
        }
    };

    v8f c[8];
#pragma unroll
    for (int i = 0; i < 8; ++i) c[i] = (v8f)(0.0f);

    issue_stage(0, 0);
    wait_async0();
    __syncthreads();

    const int nk = K >> 5;
    for (int k = 0; k < nk; ++k) {
        const int buf = k & 1;
        if (k + 1 < nk) issue_stage((k + 1) * 32, buf ^ 1);

        const _Float16* As = &Asm[buf][wm * 32][0];
        const _Float16* Bs = &Bsm[buf][wn * 64][0];
        v16h fa0 = load_frag_rm(As, LDS_LDA, 0);
        v16h fa1 = load_frag_rm(As + 16 * LDS_LDA, LDS_LDA, 0);
#pragma unroll
        for (int j = 0; j < 4; ++j) {
            v16h fb = load_frag_rm(Bs + j * 16 * LDS_LDA, LDS_LDA, 0);
            c[j]     = wmma16(fa0, fb, c[j]);
            c[4 + j] = wmma16(fa1, fb, c[4 + j]);
        }
        wait_async0();        // next stage resident before anyone flips buffers
        __syncthreads();
    }

    const int lane = threadIdx.x & 31;
    const int n = lane & 15, lh = lane >> 4;
#pragma unroll
    for (int sub = 0; sub < 2; ++sub) {
#pragma unroll
        for (int j = 0; j < 4; ++j) {
            AccU acc; acc.v = c[sub * 4 + j];
#pragma unroll
            for (int v = 0; v < 8; ++v) {
                size_t r  = (size_t)(mblk * 128 + wm * 32 + sub * 16 + v + 8 * lh);
                size_t cc = (size_t)(nblk * 128 + wn * 64 + j * 16 + n);
                if (OUTF16) ((_Float16*)Cout)[r * N + cc] = (_Float16)acc.f[v];
                else        ((float*)Cout)[r * N + cc]    = acc.f[v];
            }
        }
    }
}

// ------------------------------------------------------------ flash attention
// One wave per (head, 16-query tile). 32-key blocks, online softmax.
__global__ __launch_bounds__(128) void flash_attn(const _Float16* __restrict__ qkv,
                                                  const _Float16* __restrict__ vt,
                                                  _Float16* __restrict__ outp) {
    __shared__ __align__(16) _Float16 Plds[4][16][32];
    const int wid  = threadIdx.x >> 5;
    const int lane = threadIdx.x & 31;
    const int tile = blockIdx.x * 4 + wid;
    const int qt   = tile & 127;           // query tile (S/16 = 128)
    const int h    = tile >> 7;            // head
    const int n = lane & 15, lh = lane >> 4;

    const float scale = 0.08838834764831845f;            // 1/sqrt(128)
    const float slope = exp2f(-0.5f * (float)(h + 1));   // MPT ALiBi slope

    const _Float16* qb = qkv + (size_t)(qt * 16) * LD_QKV + h * DHEAD;
    v16h qf[4];
#pragma unroll
    for (int kk = 0; kk < 4; ++kk) qf[kk] = load_frag_rm(qb, LD_QKV, kk * 32);

    v8f o[8];
#pragma unroll
    for (int t = 0; t < 8; ++t) o[t] = (v8f)(0.0f);
    float mrow[8], lrow[8];
#pragma unroll
    for (int v = 0; v < 8; ++v) { mrow[v] = -3.0e38f; lrow[v] = 0.0f; }

    const int qmax = qt * 16 + 15;
    const int nkb  = qmax / 32 + 1;        // causal: skip fully-masked key blocks

    for (int kb = 0; kb < nkb; ++kb) {
        v8f s0 = (v8f)(0.0f), s1 = (v8f)(0.0f);
        const _Float16* kbase0 = qkv + (size_t)(kb * 32) * LD_QKV + DMODEL + h * DHEAD;
        const _Float16* kbase1 = kbase0 + (size_t)16 * LD_QKV;
#pragma unroll
        for (int kk = 0; kk < 4; ++kk) {
            v16h f0 = load_frag_rm(kbase0, LD_QKV, kk * 32);
            v16h f1 = load_frag_rm(kbase1, LD_QKV, kk * 32);
            s0 = wmma16(qf[kk], f0, s0);
            s1 = wmma16(qf[kk], f1, s1);
        }
        AccU a0, a1; a0.v = s0; a1.v = s1;

        float mnew[8], corr[8];
#pragma unroll
        for (int v = 0; v < 8; ++v) {
            int qi  = qt * 16 + v + 8 * lh;
            int k0i = kb * 32 + n;
            int k1i = k0i + 16;
            float l0 = a0.f[v] * scale + (float)(k0i - (S_LEN - 1)) * slope;
            float l1 = a1.f[v] * scale + (float)(k1i - (S_LEN - 1)) * slope;
            l0 = (k0i <= qi) ? l0 : -3.0e38f;
            l1 = (k1i <= qi) ? l1 : -3.0e38f;
            a0.f[v] = l0; a1.f[v] = l1;
            float mv = fmaxf(l0, l1);
            mv = fmaxf(mv, __shfl_xor(mv, 1));
            mv = fmaxf(mv, __shfl_xor(mv, 2));
            mv = fmaxf(mv, __shfl_xor(mv, 4));
            mv = fmaxf(mv, __shfl_xor(mv, 8));   // row lives in one lane-half
            mnew[v] = fmaxf(mrow[v], mv);
            corr[v] = __expf(mrow[v] - mnew[v]);
            mrow[v] = mnew[v];
        }

#pragma unroll
        for (int v = 0; v < 8; ++v) {
            float p0 = __expf(a0.f[v] - mnew[v]);
            float p1 = __expf(a1.f[v] - mnew[v]);
            float rs = p0 + p1;
            rs += __shfl_xor(rs, 1);
            rs += __shfl_xor(rs, 2);
            rs += __shfl_xor(rs, 4);
            rs += __shfl_xor(rs, 8);
            lrow[v] = lrow[v] * corr[v] + rs;
            Plds[wid][v + 8 * lh][n]      = (_Float16)p0;
            Plds[wid][v + 8 * lh][n + 16] = (_Float16)p1;
        }

#pragma unroll
        for (int t = 0; t < 8; ++t) {
            AccU ot; ot.v = o[t];
#pragma unroll
            for (int v = 0; v < 8; ++v) ot.f[v] *= corr[v];
            o[t] = ot.v;
        }

        v16h pf = load_frag_rm(&Plds[wid][0][0], 32, 0);
#pragma unroll
        for (int t = 0; t < 8; ++t) {
            v16h fv = load_frag_rm(vt + (size_t)(h * DHEAD + t * 16) * S_LEN, S_LEN, kb * 32);
            o[t] = wmma16(pf, fv, o[t]);
        }
    }

#pragma unroll
    for (int t = 0; t < 8; ++t) {
        AccU ot; ot.v = o[t];
#pragma unroll
        for (int v = 0; v < 8; ++v) {
            float r = ot.f[v] / lrow[v];
            size_t row = (size_t)(qt * 16 + v + 8 * lh);
            size_t col = (size_t)(h * DHEAD + t * 16 + n);
            outp[row * DMODEL + col] = (_Float16)r;
        }
    }
}

extern "C" void kernel_launch(void* const* d_in, const int* in_sizes, int n_in,
                              void* d_out, int out_size, void* d_ws, size_t ws_size,
                              hipStream_t stream) {
    const float* x     = (const float*)d_in[0];   // [S, D]
    const float* w_qkv = (const float*)d_in[1];   // [3D, D]
    const float* w_out = (const float*)d_in[2];   // [D, D]
    float* out = (float*)d_out;                   // [S, D]

    char* ws = (char*)d_ws;
    _Float16* xb    = (_Float16*)(ws);                          //  8 MB  [S,D]
    _Float16* wqkvb = (_Float16*)(ws + ((size_t)8  << 20));     // 24 MB  [3D,D]
    _Float16* woutb = (_Float16*)(ws + ((size_t)32 << 20));     //  8 MB  [D,D]
    _Float16* qkvb  = (_Float16*)(ws + ((size_t)40 << 20));     // 24 MB  [S,3D]
    _Float16* vtb   = (_Float16*)(ws + ((size_t)64 << 20));     //  8 MB  [H*d,S]
    _Float16* attnb = (_Float16*)(ws + ((size_t)72 << 20));     //  8 MB  [S,D]

    const int nX = S_LEN * DMODEL;            // 4 Mi
    const int nW = 3 * DMODEL * DMODEL;       // 12 Mi
    cvt_f32_to_f16<<<4096, 256, 0, stream>>>(x,     xb,    nX);
    cvt_f32_to_f16<<<4096, 256, 0, stream>>>(w_qkv, wqkvb, nW);
    cvt_f32_to_f16<<<4096, 256, 0, stream>>>(w_out, woutb, nX);

    // QKV GEMM: [S,D] x [3D,D]^T -> [S,3D] f16
    {
        int M = S_LEN, N = 3 * DMODEL, K = DMODEL;
        dim3 grid(N / 128, M / 128);          // 48 x 16
        gemm_lds<1><<<grid, 256, 0, stream>>>(xb, wqkvb, (void*)qkvb, M, N, K);
    }

    // V -> [H*d, S] for contiguous B-fragment loads in P x V
    transpose_v<<<(NHEADS * DHEAD * S_LEN) / 256, 256, 0, stream>>>(qkvb, vtb);

    // Flash attention: 16 heads x 128 query tiles = 2048 waves, 4 waves/block
    flash_attn<<<512, 128, 0, stream>>>(qkvb, vtb, attnb);

    // Out-proj: [S,D] x [D,D]^T -> fp32 d_out
    {
        int M = S_LEN, N = DMODEL, K = DMODEL;
        dim3 grid(N / 128, M / 128);          // 16 x 16
        gemm_lds<0><<<grid, 256, 0, stream>>>(attnb, woutb, (void*)out, M, N, K);
    }
}
